// RnnClassifier_12695923327346
// MI455X (gfx1250) — matrix operand order repaired and checked by reading, not compiled, not run on hardware
//
#include <hip/hip_runtime.h>

// ---------------------------------------------------------------------------
// Types for CDNA5 WMMA (wave32): v_wmma_f32_16x16x32_bf16
// ---------------------------------------------------------------------------
typedef __attribute__((ext_vector_type(16))) __bf16 v16bf;
typedef __attribute__((ext_vector_type(8)))  __bf16 v8bf;
typedef __attribute__((ext_vector_type(8)))  float  v8f;

union ABu { v16bf v; v8bf h[2]; };

static __device__ __forceinline__ v8f wmma_bf16(v16bf a, v16bf b, v8f c) {
    return __builtin_amdgcn_wmma_f32_16x16x32_bf16(
        /*neg_a=*/false, a, /*neg_b=*/false, b,
        /*c_mod=*/(short)0, c, /*reuse_a=*/false, /*reuse_b=*/false);
}

static __device__ __forceinline__ __bf16 f2bf(float f) {
    unsigned int u = __builtin_bit_cast(unsigned int, f);
    unsigned int r = (u + 0x7FFFu + ((u >> 16) & 1u)) >> 16;
    unsigned short s = (unsigned short)r;
    return __builtin_bit_cast(__bf16, s);
}
static __device__ __forceinline__ float bf2f(__bf16 b) {
    unsigned int u = ((unsigned int)__builtin_bit_cast(unsigned short, b)) << 16;
    return __builtin_bit_cast(float, u);
}
static __device__ __forceinline__ float sigmoidf_(float x) {
    return 1.0f / (1.0f + __expf(-x));
}

// Model dims
#define BB   256          // batch
#define TT   512          // time
#define EE   128          // embed
#define HH   256          // hidden
#define GG   768          // 3H gates
#define MM   (TT * BB)    // 131072 rows of the time-batch GEMMs

// ---------------------------------------------------------------------------
// f32 -> bf16 conversion (weights)
// ---------------------------------------------------------------------------
__global__ void cvt_bf16_kernel(const float* __restrict__ src,
                                __bf16* __restrict__ dst, int n) {
    int i = blockIdx.x * blockDim.x + threadIdx.x;
    if (i < n) dst[i] = f2bf(src[i]);
}

// ---------------------------------------------------------------------------
// Embedding gather: x[t][b][e] = emb[inputs[b][t]][e]  (bf16 out)
// ---------------------------------------------------------------------------
__global__ void embed_kernel(const int* __restrict__ inp,
                             const float* __restrict__ emb,
                             __bf16* __restrict__ x) {
    const int bx = blockIdx.x;          // = t*B + b
    const int t = bx >> 8;
    const int b = bx & 255;
    const int e = threadIdx.x;          // E threads
    const int tok = inp[b * TT + t];
    x[(size_t)bx * EE + e] = f2bf(emb[(size_t)tok * EE + e]);
}

// ---------------------------------------------------------------------------
// Input-projection GEMM (parallel, off the recurrent critical path):
//   xp[dir][t][g][b] = sum_k xin[t*B+b][k] * w_ih[dir][g][k] + b_ih[dir][g]
// Output is stored TRANSPOSED (gate-major, batch-minor) so the recurrent scan
// can fetch each lane's 8 batch rows with a single 16-byte load per gate.
// Each WG covers one timestep (256 rows = all b).  Each wave owns N-tiles
// {w, w+8, ..., w+40}; per N-tile it loads the B-fragments ONCE into
// registers and sweeps 16 M-tiles, so weight traffic = |W| per WG only.
// ---------------------------------------------------------------------------
template <int EIN>
__global__ __launch_bounds__(256) void gate_proj_kernel(
    const __bf16* __restrict__ xin,    // [M, EIN]
    const __bf16* __restrict__ w_ih,   // [2, 768, EIN]
    const float*  __restrict__ b_ih,   // [2, 768]
    __bf16* __restrict__ xp) {         // [2, T, 768, B]

    constexpr int KB = EIN / 32;       // K-chunks per fragment row
    const int dir  = blockIdx.y;
    const int tIdx = blockIdx.x;       // one timestep per WG (M = T*B, 256 rows)
    const int m0   = tIdx * 256;
    const int wave = threadIdx.x >> 5;
    const int lane = threadIdx.x & 31;
    const int n    = lane & 15;
    const int hi   = lane >> 4;

    const __bf16* Wih = w_ih + (size_t)dir * GG * EIN;
    const float*  bih = b_ih + dir * GG;
    __bf16* xpo = xp + ((size_t)dir * TT + tIdx) * GG * BB;

#pragma unroll 1
    for (int nt = wave; nt < GG / 16; nt += 8) {
        const int col = nt * 16 + n;
        // ---- load this N-tile's B fragments once --------------------------
        v16bf bw[KB];
        const __bf16* Brow = Wih + (size_t)col * EIN;
#pragma unroll
        for (int kk = 0; kk < KB; ++kk)
            bw[kk] = *(const v16bf*)(Brow + kk * 32 + hi * 16);
        const float bias = bih[col];

        // ---- sweep 16 M-tiles reusing the registered weights --------------
#pragma unroll 1
        for (int mt = 0; mt < 16; ++mt) {
            const __bf16* arow = xin + (size_t)(m0 + mt * 16 + n) * EIN;
            v8f acc = {};
#pragma unroll
            for (int kk = 0; kk < KB; ++kk) {
                ABu a;
                a.h[0] = *(const v8bf*)(arow + kk * 32 + hi * 16);
                a.h[1] = *(const v8bf*)(arow + kk * 32 + hi * 16 + 8);
                acc = wmma_bf16(a.v, bw[kk], acc);
            }
#pragma unroll
            for (int j = 0; j < 8; ++j) {
                const int b = mt * 16 + j + 8 * hi;   // batch index
                xpo[(size_t)col * BB + b] = f2bf(acc[j] + bias);
            }
        }
    }
}

// ---------------------------------------------------------------------------
// Recurrent-only GRU scan.  Batch tile of 16 rows per blockIdx.x, direction
// per blockIdx.y.  The w_hh fragments this wave needs (3 gate tiles x 8
// K-chunks = 24 fragments = 192 VGPRs) are loaded ONCE and persist across all
// 512 steps; the only per-step global traffic is three 16B loads per lane of
// the transposed precomputed gates.  h lives in LDS as bf16.
//   xp : [2, T, 768, B] bf16 (includes b_ih)  w_hh: [2, 768, 256] bf16
//   out: [T, B, 512] bf16 (dir 0 -> cols [0,256), dir 1 -> [256,512))
// ---------------------------------------------------------------------------
__global__ __launch_bounds__(512) void gru_scan_kernel(
    const __bf16* __restrict__ xp,
    const __bf16* __restrict__ w_hh,
    const float*  __restrict__ b_hh,
    __bf16* __restrict__ out) {

    const int dir  = blockIdx.y;
    const int b0   = blockIdx.x * 16;
    const int wave = threadIdx.x >> 5;
    const int lane = threadIdx.x & 31;
    const int n    = lane & 15;
    const int hi   = lane >> 4;

    const __bf16* Whh = w_hh + (size_t)dir * GG * HH;
    const float*  bhh = b_hh + dir * GG;
    const __bf16* xpd = xp + (size_t)dir * TT * GG * BB;

    const int jcol = wave * 16 + n;                 // output column in [0,256)
    const float bhr = bhh[jcol], bhz = bhh[256 + jcol], bhn = bhh[512 + jcol];

    // ---- persist w_hh B-fragments in VGPRs for the whole scan ------------
    v16bf whr[8], whz[8], whn[8];
    {
        const __bf16* Br = Whh + (size_t)(      jcol) * HH;
        const __bf16* Bz = Whh + (size_t)(256 + jcol) * HH;
        const __bf16* Bn = Whh + (size_t)(512 + jcol) * HH;
#pragma unroll
        for (int kk = 0; kk < 8; ++kk) {
            whr[kk] = *(const v16bf*)(Br + kk * 32 + hi * 16);
            whz[kk] = *(const v16bf*)(Bz + kk * 32 + hi * 16);
            whn[kk] = *(const v16bf*)(Bn + kk * 32 + hi * 16);
        }
    }

    // per-lane offset of its 8 contiguous batch values in the xp slice
    const size_t gofs_r = (size_t)(      jcol) * BB + b0 + 8 * hi;
    const size_t gofs_z = (size_t)(256 + jcol) * BB + b0 + 8 * hi;
    const size_t gofs_n = (size_t)(512 + jcol) * BB + b0 + 8 * hi;

    __shared__ alignas(16) __bf16 h_lds[16 * HH];
    for (int i = threadIdx.x; i < 16 * HH; i += blockDim.x)
        h_lds[i] = __builtin_bit_cast(__bf16, (unsigned short)0);
    __syncthreads();

    for (int t = 0; t < TT; ++t) {
        const int tt = dir ? (TT - 1 - t) : t;
        const __bf16* gslice = xpd + (size_t)tt * GG * BB;

        // ---- fetch precomputed gates: 3 x 16B per lane (overlaps WMMA) ---
        const v8bf gr = *(const v8bf*)(gslice + gofs_r);
        const v8bf gz = *(const v8bf*)(gslice + gofs_z);
        const v8bf gn = *(const v8bf*)(gslice + gofs_n);
        {   // prefetch next timestep's gate slice (global_prefetch_b8)
            const int tn = dir ? (tt > 0 ? tt - 1 : tt) : (tt + 1 < TT ? tt + 1 : tt);
            __builtin_prefetch(xpd + (size_t)tn * GG * BB + gofs_r, 0, 1);
        }

        // ---- recurrent projection: Gh = h @ Whh^T (A from LDS) -----------
        v8f ah_r = {}, ah_z = {}, ah_n = {};
        const __bf16* hrow = h_lds + n * HH;
#pragma unroll
        for (int kk = 0; kk < 8; ++kk) {
            ABu a;
            a.h[0] = *(const v8bf*)(hrow + kk * 32 + hi * 16);
            a.h[1] = *(const v8bf*)(hrow + kk * 32 + hi * 16 + 8);
            ah_r = wmma_bf16(a.v, whr[kk], ah_r);
            ah_z = wmma_bf16(a.v, whz[kk], ah_z);
            ah_n = wmma_bf16(a.v, whn[kk], ah_n);
        }

        __syncthreads();   // all waves done reading h_lds

        // ---- GRU gate combine entirely in registers ----------------------
#pragma unroll
        for (int j = 0; j < 8; ++j) {
            const int m = j + 8 * hi;  // batch row within tile
            float r  = sigmoidf_(bf2f(gr[j]) + ah_r[j] + bhr);
            float z  = sigmoidf_(bf2f(gz[j]) + ah_z[j] + bhz);
            float nn = tanhf(bf2f(gn[j]) + r * (ah_n[j] + bhn));
            float ho = bf2f(h_lds[m * HH + jcol]);
            float hv = (1.0f - z) * nn + z * ho;
            h_lds[m * HH + jcol] = f2bf(hv);
        }

        __syncthreads();   // new h visible to everyone

        // ---- stream h_t to global out (coalesced 128-bit stores) ---------
        {
            const int i   = threadIdx.x * 8;    // 512 thr * 8 = 4096 = 16*256
            const int row = i >> 8;
            const int col = i & 255;
            *(v8bf*)(out + ((size_t)tt * BB + b0 + row) * 512 + dir * HH + col) =
                *(const v8bf*)(h_lds + i);
        }
        // copy's reads are protected by next iteration's first __syncthreads
    }
}

// ---------------------------------------------------------------------------
// FC1: fc1[b,n] = leaky_relu( sum_k flat[b,k]*W1[n,k] + b1[n] )
// flat[b, t*512+c] == h2[t][b][c]  (read in place, no materialized transpose)
// ---------------------------------------------------------------------------
__global__ __launch_bounds__(256) void fc1_kernel(
    const __bf16* __restrict__ h2,
    const __bf16* __restrict__ W1,
    const float*  __restrict__ b1,
    float* __restrict__ fc1) {

    const int wave = threadIdx.x >> 5;
    const int lane = threadIdx.x & 31;
    const int n    = lane & 15;
    const int hi   = lane >> 4;
    const int b0   = blockIdx.x * 16;
    const int n0   = wave * 16;

    const __bf16* Brow = W1 + (size_t)(n0 + n) * (TT * 512);
    const size_t  bRow = b0 + n;

    v8f acc = {};
    for (int kk = 0; kk < (TT * 512) / 32; ++kk) {
        const int k0    = kk * 32;
        const int tblk  = k0 >> 9;
        const int cbase = k0 & 511;
        const __bf16* arow = h2 + ((size_t)tblk * BB + bRow) * 512 + cbase;
        ABu a;
        a.h[0] = *(const v8bf*)(arow + hi * 16);
        a.h[1] = *(const v8bf*)(arow + hi * 16 + 8);
        v16bf bb = *(const v16bf*)(Brow + k0 + hi * 16);
        acc = wmma_bf16(a.v, bb, acc);
    }

#pragma unroll
    for (int j = 0; j < 8; ++j) {
        const int row = b0 + j + 8 * hi;
        const int col = n0 + n;
        float v = acc[j] + b1[col];
        v = v > 0.0f ? v : 0.01f * v;    // leaky_relu
        fc1[row * 128 + col] = v;
    }
}

// ---------------------------------------------------------------------------
// FC2 + sigmoid
// ---------------------------------------------------------------------------
__global__ void fc2_kernel(const float* __restrict__ fc1,
                           const float* __restrict__ W2,
                           const float* __restrict__ b2,
                           float* __restrict__ outp) {
    const int b = blockIdx.x * blockDim.x + threadIdx.x;
    if (b >= BB) return;
    float s = b2[0];
#pragma unroll 8
    for (int i = 0; i < 128; ++i) s += fc1[b * 128 + i] * W2[i];
    outp[b] = sigmoidf_(s);
}

// ---------------------------------------------------------------------------
// Launcher
// ---------------------------------------------------------------------------
extern "C" void kernel_launch(void* const* d_in, const int* in_sizes, int n_in,
                              void* d_out, int out_size, void* d_ws, size_t ws_size,
                              hipStream_t stream) {
    (void)in_sizes; (void)n_in; (void)out_size; (void)ws_size;

    const int*   inputs  = (const int*)  d_in[0];
    const float* emb     = (const float*)d_in[1];
    const float* w_ih_l0 = (const float*)d_in[2];
    const float* w_hh_l0 = (const float*)d_in[3];
    const float* b_ih_l0 = (const float*)d_in[4];
    const float* b_hh_l0 = (const float*)d_in[5];
    const float* w_ih_l1 = (const float*)d_in[6];
    const float* w_hh_l1 = (const float*)d_in[7];
    const float* b_ih_l1 = (const float*)d_in[8];
    const float* b_hh_l1 = (const float*)d_in[9];
    const float* W1      = (const float*)d_in[10];
    const float* b1      = (const float*)d_in[11];
    const float* W2      = (const float*)d_in[12];
    const float* b2      = (const float*)d_in[13];
    float* outp = (float*)d_out;                    // [B,1] f32

    // ---- workspace carve-up ----------------------------------------------
    char* ws = (char*)d_ws;
    size_t off = 0;
    auto take = [&](size_t bytes) { char* p = ws + off; off += (bytes + 255) & ~(size_t)255; return p; };

    __bf16* x_bf  = (__bf16*)take((size_t)MM * EE * 2);            // 33.5 MB
    __bf16* wih0  = (__bf16*)take((size_t)2 * GG * EE * 2);
    __bf16* whh0  = (__bf16*)take((size_t)2 * GG * HH * 2);
    __bf16* wih1  = (__bf16*)take((size_t)2 * GG * 512 * 2);
    __bf16* whh1  = (__bf16*)take((size_t)2 * GG * HH * 2);
    __bf16* W1bf  = (__bf16*)take((size_t)128 * TT * 512 * 2);     // 67 MB
    __bf16* h1    = (__bf16*)take((size_t)MM * 512 * 2);           // 134 MB
    __bf16* h2    = (__bf16*)take((size_t)MM * 512 * 2);           // 134 MB
    __bf16* xp    = (__bf16*)take((size_t)2 * MM * GG * 2);        // 403 MB (shared by both layers)
    float*  fc1   = (float*) take((size_t)BB * 128 * 4);

    // ---- weight conversions f32 -> bf16 ----------------------------------
    auto cvt = [&](const float* s, __bf16* d, int nelem) {
        cvt_bf16_kernel<<<(nelem + 255) / 256, 256, 0, stream>>>(s, d, nelem);
    };
    cvt(w_ih_l0, wih0, 2 * GG * EE);
    cvt(w_hh_l0, whh0, 2 * GG * HH);
    cvt(w_ih_l1, wih1, 2 * GG * 512);
    cvt(w_hh_l1, whh1, 2 * GG * HH);
    cvt(W1,      W1bf, 128 * TT * 512);

    // ---- embedding gather ------------------------------------------------
    embed_kernel<<<MM, EE, 0, stream>>>(inputs, emb, x_bf);

    dim3 pgrid(TT, 2);             // input-projection GEMMs (one WG per t)
    dim3 sgrid(BB / 16, 2);        // recurrent scans (sequential over T)

    // ---- layer 0 ---------------------------------------------------------
    gate_proj_kernel<EE><<<pgrid, 256, 0, stream>>>(x_bf, wih0, b_ih_l0, xp);
    gru_scan_kernel<<<sgrid, 512, 0, stream>>>(xp, whh0, b_hh_l0, h1);

    // ---- layer 1 ---------------------------------------------------------
    gate_proj_kernel<512><<<pgrid, 256, 0, stream>>>(h1, wih1, b_ih_l1, xp);
    gru_scan_kernel<<<sgrid, 512, 0, stream>>>(xp, whh1, b_hh_l1, h2);

    // ---- FC head ---------------------------------------------------------
    fc1_kernel<<<BB / 16, 256, 0, stream>>>(h2, W1bf, b1, fc1);
    fc2_kernel<<<1, 256, 0, stream>>>(fc1, W2, b2, outp);
}
